// GatedAttentionUnit_20950850470540
// MI455X (gfx1250) — matrix operand order repaired and checked by reading, not compiled, not run on hardware
//
#include <hip/hip_runtime.h>
#include <hip/hip_bf16.h>

typedef __attribute__((ext_vector_type(16))) _Float16 v16h;
typedef __attribute__((ext_vector_type(8)))  _Float16 v8h;
typedef __attribute__((ext_vector_type(8)))  float    v8f;
typedef __attribute__((ext_vector_type(4)))  unsigned v4u;
typedef __attribute__((ext_vector_type(8)))  unsigned v8u;

// Problem constants
#define BATCH 8
#define LSEQ  2048
#define DDIM  512
#define SDIM  128
#define EDIM  1024          // 2*D
#define NCOLS 2176          // 2*E + S
#define BLROWS (BATCH*LSEQ) // 16384

// GEMM tiling
#define BM 128
#define BN 128
#define BK 32
#define LSTR 48                 // LDS row stride in halves (96B, 16B-aligned)
#define TILE_H (BM * LSTR)      // halves per LDS tile buffer

// ---------------------------------------------------------------------------
// CDNA5 async global->LDS copy (ASYNCcnt path), 32B per lane  (prologue tiles)
// ---------------------------------------------------------------------------
__device__ inline void async_copy_32B(const _Float16* __restrict__ g, _Float16* l) {
    unsigned lds = (unsigned)(size_t)l;   // low 32 bits of generic addr = LDS offset
    asm volatile(
        "global_load_async_to_lds_b128 %0, %1, off\n\t"
        "global_load_async_to_lds_b128 %0, %1, off offset:16"
        :: "v"(lds), "v"(g) : "memory");
}
__device__ inline void wait_async() {
    asm volatile("s_wait_asynccnt 0" ::: "memory");
}

__device__ inline void stage_a(const _Float16* __restrict__ A, int lda,
                               int m0, int k0, _Float16* lA, int tid) {
    int row = tid >> 1;
    int col = (tid & 1) * 16;
    async_copy_32B(A + (size_t)(m0 + row) * lda + k0 + col, lA + row * LSTR + col);
}
__device__ inline void stage_bt(const _Float16* __restrict__ BT, int ldbt,
                                int n0, int k0, _Float16* lBT, int tid) {
    int n = tid >> 1;
    int col = (tid & 1) * 16;
    async_copy_32B(BT + (size_t)(n0 + n) * ldbt + k0 + col, lBT + n * LSTR + col);
}

// ---------------------------------------------------------------------------
// Tensor Data Mover: one descriptor moves a 128-row x 64B tile (row stride
// ld halves) into LDS with 32B padding per row (LDS row stride = 96B).
//   data_size=2B, tile_dim0=32, tile_dim1=128, pad_interval=3 (16 DW = 64B),
//   pad_amount=7 (8 DW = 32B), type=2 ("image"), count=1.
// Issued by one wave only; tracked with TENSORcnt.
// ---------------------------------------------------------------------------
__device__ inline void tdm_load_tile(const _Float16* __restrict__ g, int ld,
                                     _Float16* lds_tile) {
    unsigned lds = (unsigned)(size_t)lds_tile;
    unsigned long long ga = (unsigned long long)(size_t)g;
    v4u g0;
    g0[0] = 1u;                                   // count=1, user mode, no gather
    g0[1] = lds;                                  // lds_addr
    g0[2] = (unsigned)ga;                         // global_addr[31:0]
    g0[3] = (unsigned)(ga >> 32) | (2u << 30);    // global_addr[56:32] | type=2
    v8u g1;
    g1[0] = (1u << 16)        // data_size = 2 bytes
          | (1u << 20)        // pad_enable
          | (3u << 22)        // pad_interval: 16 DWORDs (=64B row)
          | (7u << 25);       // pad_amount:   8 DWORDs (=32B pad)
    g1[1] = 0xFFFF0000u;                          // tensor_dim0[15:0]  (huge, no OOB)
    g1[2] = 0x7FFFu | (0xFFFFu << 16);            // tensor_dim0[31:16] | tensor_dim1[15:0]
    g1[3] = 0x7FFFu | (32u << 16);                // tensor_dim1[31:16] | tile_dim0=32
    g1[4] = 128u;                                 // tile_dim1=128, tile_dim2=0
    g1[5] = (unsigned)ld;                         // tensor_dim0_stride (elements)
    g1[6] = 0u;
    g1[7] = 0u;
    asm volatile("tensor_load_to_lds %0, %1" :: "s"(g0), "s"(g1) : "memory");
}

// ---------------------------------------------------------------------------
// One K-step of WMMA for a wave: 32 rows x 64 cols (2 M-subtiles x 4 N-subtiles).
// All fragments loaded first (sched_barrier keeps the group together), then
// 8 WMMAs issue back-to-back with full A/B fragment reuse.
// ---------------------------------------------------------------------------
__device__ inline void wave_mma_step(const _Float16* lA, const _Float16* lBT,
                                     int wm, int wn, int lane, v8f acc[2][4]) {
    const int m  = lane & 15;
    const int hi = lane >> 4;
    v16h a[2];
#pragma unroll
    for (int mi = 0; mi < 2; ++mi) {
        const _Float16* pa = lA + (wm + mi * 16 + m) * LSTR + hi * 8;
        v8h a0 = *(const v8h*)pa;          // K 0..7   (or 8..15)
        v8h a1 = *(const v8h*)(pa + 16);   // K 16..23 (or 24..31)
#pragma unroll
        for (int i = 0; i < 8; ++i) { a[mi][i] = a0[i]; a[mi][8 + i] = a1[i]; }
    }
    v16h b[4];
#pragma unroll
    for (int ns = 0; ns < 4; ++ns) {
        const _Float16* pb = lBT + (wn + ns * 16 + m) * LSTR + hi * 8;
        v8h b0 = *(const v8h*)pb;
        v8h b1 = *(const v8h*)(pb + 16);
#pragma unroll
        for (int i = 0; i < 8; ++i) { b[ns][i] = b0[i]; b[ns][8 + i] = b1[i]; }
    }
    __builtin_amdgcn_sched_barrier(0);     // loads above, WMMAs below
#pragma unroll
    for (int ns = 0; ns < 4; ++ns)
#pragma unroll
        for (int mi = 0; mi < 2; ++mi)
            acc[mi][ns] = __builtin_amdgcn_wmma_f32_16x16x32_f16(
                false, a[mi], false, b[ns], (short)0, acc[mi][ns], false, false);
}

// ---------------------------------------------------------------------------
// Double-buffered GEMM mainloop shared by all GEMM kernels.
// Prologue tile via per-lane async copies; steady state via TDM descriptors.
// ---------------------------------------------------------------------------
__device__ inline void gemm_main(const _Float16* __restrict__ A, int lda,
                                 const _Float16* __restrict__ BT, int ldbt,
                                 int m0, int n0, int K,
                                 _Float16* lA, _Float16* lBT,
                                 int tid, int lane, int wm, int wn, v8f acc[2][4]) {
    const int wave = tid >> 5;
    stage_a(A, lda, m0, 0, lA, tid);
    stage_bt(BT, ldbt, n0, 0, lBT, tid);
    wait_async();
    __syncthreads();
    int buf = 0;
    for (int k0 = 0; k0 < K; k0 += BK) {
        const int nbuf = buf ^ 1;
        if (k0 + BK < K && wave == 0) {
            tdm_load_tile(A + (size_t)m0 * lda + (k0 + BK), lda, lA + nbuf * TILE_H);
            tdm_load_tile(BT + (size_t)n0 * ldbt + (k0 + BK), ldbt, lBT + nbuf * TILE_H);
        }
        wave_mma_step(lA + buf * TILE_H, lBT + buf * TILE_H, wm, wn, lane, acc);
        if (wave == 0) __builtin_amdgcn_s_wait_tensorcnt(0);
        __syncthreads();
        buf = nbuf;
    }
}

// ---------------------------------------------------------------------------
// Kernel: transpose + f32->f16 convert (for W_uv, W_o).  src R x C -> dst C x R
// ---------------------------------------------------------------------------
__global__ void k_transpose_f32_f16(const float* __restrict__ src, _Float16* __restrict__ dst,
                                    int R, int C) {
    __shared__ float t[32][33];
    const int c0 = blockIdx.x * 32, r0 = blockIdx.y * 32;
    const int tx = threadIdx.x & 31, ty = threadIdx.x >> 5;   // 32 x 8
#pragma unroll
    for (int i = 0; i < 32; i += 8)
        t[ty + i][tx] = src[(size_t)(r0 + ty + i) * C + c0 + tx];
    __syncthreads();
#pragma unroll
    for (int i = 0; i < 32; i += 8)
        dst[(size_t)(c0 + ty + i) * R + r0 + tx] = (_Float16)t[tx][ty + i];
}

// f16 -> f16 transpose, batched over blockIdx.z (for V -> Vt)
__global__ void k_transpose_f16(const _Float16* __restrict__ src, _Float16* __restrict__ dst,
                                int R, int C) {
    __shared__ _Float16 t[32][34];
    const size_t zoff = (size_t)blockIdx.z * R * C;
    const int c0 = blockIdx.x * 32, r0 = blockIdx.y * 32;
    const int tx = threadIdx.x & 31, ty = threadIdx.x >> 5;
#pragma unroll
    for (int i = 0; i < 32; i += 8)
        t[ty + i][tx] = src[zoff + (size_t)(r0 + ty + i) * C + c0 + tx];
    __syncthreads();
#pragma unroll
    for (int i = 0; i < 32; i += 8)
        dst[zoff + (size_t)(c0 + ty + i) * R + r0 + tx] = t[tx][ty + i];
}

// ---------------------------------------------------------------------------
// Kernel: RMS norm -> xn (f16). One block (256 thr) per row of 512.
// ---------------------------------------------------------------------------
__global__ void k_rmsnorm(const float* __restrict__ x, const float* __restrict__ g_ln,
                          _Float16* __restrict__ Xn) {
    const int row  = blockIdx.x;
    const int t    = threadIdx.x;
    const int lane = t & 31, wave = t >> 5;
    const float v0 = x[(size_t)row * DDIM + t];
    const float v1 = x[(size_t)row * DDIM + 256 + t];
    float ss = v0 * v0 + v1 * v1;
#pragma unroll
    for (int o = 16; o > 0; o >>= 1) ss += __shfl_xor(ss, o, 32);
    __shared__ float red[8];
    if (lane == 0) red[wave] = ss;
    __syncthreads();
    float tot = 0.f;
#pragma unroll
    for (int i = 0; i < 8; ++i) tot += red[i];
    const float norm  = sqrtf(tot * (1.0f / (float)DDIM));
    const float scale = g_ln[0] / fmaxf(norm, 1e-5f);
    Xn[(size_t)row * DDIM + t]       = (_Float16)(v0 * scale);
    Xn[(size_t)row * DDIM + 256 + t] = (_Float16)(v1 * scale);
}

// ---------------------------------------------------------------------------
// Kernel: uv = silu(Xn @ Wuv); split into U, V, Q(=g0*b+b0), K(=g1*b+b1)
// B operand pre-transposed: WuvT[N=2176][K=512]
// ---------------------------------------------------------------------------
__global__ void k_gemm_uv(const _Float16* __restrict__ Xn, const _Float16* __restrict__ WuvT,
                          const float* __restrict__ gamma, const float* __restrict__ beta,
                          _Float16* __restrict__ U, _Float16* __restrict__ V,
                          _Float16* __restrict__ Qh, _Float16* __restrict__ Kh) {
    __shared__ _Float16 lA[2 * TILE_H];
    __shared__ _Float16 lBT[2 * TILE_H];
    const int tid = threadIdx.x, lane = tid & 31, wave = tid >> 5;
    const int wm = (wave & 3) * 32, wn = (wave >> 2) * 64;
    const int m0 = blockIdx.y * BM, n0 = blockIdx.x * BN;
    v8f acc[2][4] = {};
    gemm_main(Xn, DDIM, WuvT, DDIM, m0, n0, DDIM, lA, lBT, tid, lane, wm, wn, acc);
    const int m = lane & 15, hi = lane >> 4;
#pragma unroll
    for (int mi = 0; mi < 2; ++mi)
#pragma unroll
    for (int ns = 0; ns < 4; ++ns) {
        const int c = n0 + wn + ns * 16 + m;
#pragma unroll
        for (int r = 0; r < 8; ++r) {
            const int row = m0 + wm + mi * 16 + hi * 8 + r;
            const float a = acc[mi][ns][r];
            const float s = a / (1.0f + __expf(-a));   // SiLU
            if (c < EDIM) {
                U[(size_t)row * EDIM + c] = (_Float16)s;
            } else if (c < 2 * EDIM) {
                V[(size_t)row * EDIM + (c - EDIM)] = (_Float16)s;
            } else {
                const int si = c - 2 * EDIM;
                Qh[(size_t)row * SDIM + si] = (_Float16)(s * gamma[si] + beta[si]);
                Kh[(size_t)row * SDIM + si] = (_Float16)(s * gamma[SDIM + si] + beta[SDIM + si]);
            }
        }
    }
}

// ---------------------------------------------------------------------------
// Kernel: scores = relu((Q K^T + relbias)/sqrt(S))^2   (per batch, f16 out)
// ---------------------------------------------------------------------------
__global__ void k_scores(const _Float16* __restrict__ Qh, const _Float16* __restrict__ Kh,
                         const float* __restrict__ w_rel, _Float16* __restrict__ Sc) {
    __shared__ _Float16 lA[2 * TILE_H];
    __shared__ _Float16 lBT[2 * TILE_H];
    const int tid = threadIdx.x, lane = tid & 31, wave = tid >> 5;
    const int wm = (wave & 3) * 32, wn = (wave >> 2) * 64;
    const int z = blockIdx.z;
    const _Float16* Qb = Qh + (size_t)z * LSEQ * SDIM;
    const _Float16* Kb = Kh + (size_t)z * LSEQ * SDIM;
    _Float16* Sb = Sc + (size_t)z * LSEQ * LSEQ;
    const int m0 = blockIdx.y * BM, n0 = blockIdx.x * BN;
    v8f acc[2][4] = {};
    gemm_main(Qb, SDIM, Kb, SDIM, m0, n0, SDIM, lA, lBT, tid, lane, wm, wn, acc);
    const int m = lane & 15, hi = lane >> 4;
    const float inv_sqrt_s = 0.08838834764831845f;   // 1/sqrt(128)
#pragma unroll
    for (int mi = 0; mi < 2; ++mi)
#pragma unroll
    for (int ns = 0; ns < 4; ++ns) {
        const int j = n0 + wn + ns * 16 + m;
#pragma unroll
        for (int r = 0; r < 8; ++r) {
            const int i = m0 + wm + mi * 16 + hi * 8 + r;
            const float bias = w_rel[(LSEQ - 1) + j - i];
            const float v = (acc[mi][ns][r] + bias) * inv_sqrt_s;
            const float rl = fmaxf(v, 0.0f);
            Sb[(size_t)i * LSEQ + j] = (_Float16)(rl * rl);
        }
    }
}

// ---------------------------------------------------------------------------
// Kernel: H = U .* (scores @ V)   (per batch, f16 out); B operand = Vt[E][L]
// ---------------------------------------------------------------------------
__global__ void k_av(const _Float16* __restrict__ Sc, const _Float16* __restrict__ Vt,
                     const _Float16* __restrict__ U, _Float16* __restrict__ H) {
    __shared__ _Float16 lA[2 * TILE_H];
    __shared__ _Float16 lBT[2 * TILE_H];
    const int tid = threadIdx.x, lane = tid & 31, wave = tid >> 5;
    const int wm = (wave & 3) * 32, wn = (wave >> 2) * 64;
    const int z = blockIdx.z;
    const _Float16* Ab  = Sc + (size_t)z * LSEQ * LSEQ;
    const _Float16* Vtb = Vt + (size_t)z * EDIM * LSEQ;
    const int m0 = blockIdx.y * BM, n0 = blockIdx.x * BN;
    v8f acc[2][4] = {};
    gemm_main(Ab, LSEQ, Vtb, LSEQ, m0, n0, LSEQ, lA, lBT, tid, lane, wm, wn, acc);
    const int m = lane & 15, hi = lane >> 4;
#pragma unroll
    for (int mi = 0; mi < 2; ++mi)
#pragma unroll
    for (int ns = 0; ns < 4; ++ns) {
        const int col = n0 + wn + ns * 16 + m;
#pragma unroll
        for (int r = 0; r < 8; ++r) {
            const int grow = z * LSEQ + m0 + wm + mi * 16 + hi * 8 + r;
            const float u = (float)U[(size_t)grow * EDIM + col];
            H[(size_t)grow * EDIM + col] = (_Float16)(acc[mi][ns][r] * u);
        }
    }
}

// ---------------------------------------------------------------------------
// Kernel: out = x*res_scale + H @ Wo   (f32 out); B operand = WoT[512][1024]
// ---------------------------------------------------------------------------
__global__ void k_out(const _Float16* __restrict__ H, const _Float16* __restrict__ WoT,
                      const float* __restrict__ x, const float* __restrict__ res_scale,
                      float* __restrict__ out) {
    __shared__ _Float16 lA[2 * TILE_H];
    __shared__ _Float16 lBT[2 * TILE_H];
    const int tid = threadIdx.x, lane = tid & 31, wave = tid >> 5;
    const int wm = (wave & 3) * 32, wn = (wave >> 2) * 64;
    const int m0 = blockIdx.y * BM, n0 = blockIdx.x * BN;
    v8f acc[2][4] = {};
    gemm_main(H, EDIM, WoT, EDIM, m0, n0, EDIM, lA, lBT, tid, lane, wm, wn, acc);
    const int m = lane & 15, hi = lane >> 4;
#pragma unroll
    for (int mi = 0; mi < 2; ++mi)
#pragma unroll
    for (int ns = 0; ns < 4; ++ns) {
        const int col = n0 + wn + ns * 16 + m;
#pragma unroll
        for (int r = 0; r < 8; ++r) {
            const int row = m0 + wm + mi * 16 + hi * 8 + r;
            out[(size_t)row * DDIM + col] =
                x[(size_t)row * DDIM + col] * res_scale[col] + acc[mi][ns][r];
        }
    }
}

// ---------------------------------------------------------------------------
// Launcher
// ---------------------------------------------------------------------------
extern "C" void kernel_launch(void* const* d_in, const int* in_sizes, int n_in,
                              void* d_out, int out_size, void* d_ws, size_t ws_size,
                              hipStream_t stream) {
    const float* x         = (const float*)d_in[0];
    const float* W_uv      = (const float*)d_in[1];
    const float* gamma     = (const float*)d_in[2];
    const float* beta      = (const float*)d_in[3];
    const float* w_rel     = (const float*)d_in[4];
    const float* W_o       = (const float*)d_in[5];
    const float* g_ln      = (const float*)d_in[6];
    const float* res_scale = (const float*)d_in[7];
    float* out = (float*)d_out;

    // Workspace carve-up (all f16)
    _Float16* ws = (_Float16*)d_ws;
    size_t o = 0;
    _Float16* WuvT = ws + o; o += (size_t)NCOLS * DDIM;       // 2176 x 512
    _Float16* WoT  = ws + o; o += (size_t)DDIM * EDIM;        // 512 x 1024
    _Float16* Xn   = ws + o; o += (size_t)BLROWS * DDIM;
    _Float16* U    = ws + o; o += (size_t)BLROWS * EDIM;
    _Float16* V    = ws + o; o += (size_t)BLROWS * EDIM;
    _Float16* Vt   = ws + o; o += (size_t)BLROWS * EDIM;      // per batch: E x L
    _Float16* Qh   = ws + o; o += (size_t)BLROWS * SDIM;
    _Float16* Kh   = ws + o; o += (size_t)BLROWS * SDIM;
    _Float16* Sc   = ws + o; o += (size_t)BATCH * LSEQ * LSEQ;
    _Float16* H    = ws + o; o += (size_t)BLROWS * EDIM;

    // Pre-transpose (and convert) all GEMM B operands
    k_transpose_f32_f16<<<dim3(NCOLS / 32, DDIM / 32), 256, 0, stream>>>(W_uv, WuvT, DDIM, NCOLS);
    k_transpose_f32_f16<<<dim3(DDIM / 32, EDIM / 32), 256, 0, stream>>>(W_o, WoT, EDIM, DDIM);

    k_rmsnorm<<<BLROWS, 256, 0, stream>>>(x, g_ln, Xn);

    k_gemm_uv<<<dim3(NCOLS / BN, BLROWS / BM), 256, 0, stream>>>(
        Xn, WuvT, gamma, beta, U, V, Qh, Kh);

    k_transpose_f16<<<dim3(EDIM / 32, LSEQ / 32, BATCH), 256, 0, stream>>>(V, Vt, LSEQ, EDIM);

    k_scores<<<dim3(LSEQ / BN, LSEQ / BM, BATCH), 256, 0, stream>>>(Qh, Kh, w_rel, Sc);

    k_av<<<dim3(EDIM / BN, LSEQ / BM, BATCH), 256, 0, stream>>>(Sc, Vt, U, H);

    k_out<<<dim3(DDIM / BN, BLROWS / BM), 256, 0, stream>>>(H, WoT, x, res_scale, out);
}
